// RPEMultiheadSelfAttention_89154931130720
// MI455X (gfx1250) — matrix operand order repaired and checked by reading, not compiled, not run on hardware
//
#include <hip/hip_runtime.h>

// ---------------------------------------------------------------------------
// RPE multi-head self attention for MI455X (gfx1250, wave32, WMMA).
//  K0: pack x / w_qkv f32 -> bf16 (native cvt, vector loads/stores)
//  K0b: pack w_rpe -> padded [64 buckets][64 d] bf16 B-layout buffer
//  K1: qkv = x @ w_qkv^T  (v_wmma_f32_16x16x32_bf16), Q scaled, V transposed
//  K2: fused flash attention; RPE bias table via WMMA; bucket LUT in LDS;
//      all K/V/P fragments are contiguous 16/32-byte loads.
// ---------------------------------------------------------------------------

typedef __attribute__((ext_vector_type(8)))  __bf16 v8bf;
typedef __attribute__((ext_vector_type(16))) __bf16 v16bf;
typedef __attribute__((ext_vector_type(8)))  float  v8f;

#define L_SEQ   3136
#define IN_DIMC 512
#define ATT_DIM 512
#define NH      8
#define HD      64
#define NB      50      // rpe buckets (7*7+1)
#define NBP     64      // padded buckets for WMMA
#define SIDE    56      // sqrt(L)

struct bfpair { v8bf lo, hi; };

__device__ __forceinline__ v16bf fuse(v8bf lo, v8bf hi) {
    bfpair p{lo, hi};
    return __builtin_bit_cast(v16bf, p);
}

// ---------------------------------------------------------------------------
// K0: f32 -> bf16 pack, 8 elements/thread (packed cvt + b128 store)
// ---------------------------------------------------------------------------
__global__ __launch_bounds__(256) void pack_bf16(
    const float* __restrict__ in, __bf16* __restrict__ out, int n8)
{
    int i = blockIdx.x * 256 + threadIdx.x;
    if (i >= n8) return;
    const float4* p = (const float4*)(in + (size_t)i * 8);
    float4 a = p[0], b = p[1];
    v8bf o;
    o[0] = (__bf16)a.x; o[1] = (__bf16)a.y; o[2] = (__bf16)a.z; o[3] = (__bf16)a.w;
    o[4] = (__bf16)b.x; o[5] = (__bf16)b.y; o[6] = (__bf16)b.z; o[7] = (__bf16)b.w;
    *(v8bf*)(out + (size_t)i * 8) = o;
}

// ---------------------------------------------------------------------------
// K0b: w_rpe (64,50) f32 -> wrt [n=64 padded][d=64] bf16 (B-fragment friendly)
// ---------------------------------------------------------------------------
__global__ __launch_bounds__(256) void pack_wrpe(
    const float* __restrict__ w_rpe, __bf16* __restrict__ wrt)
{
    int i = blockIdx.x * 256 + threadIdx.x;
    if (i >= NBP * HD) return;
    int n = i / HD, d = i % HD;
    wrt[n * HD + d] = (n < NB) ? (__bf16)w_rpe[d * NB + n] : (__bf16)0.0f;
}

// ---------------------------------------------------------------------------
// K1: QKV GEMM.  M=3136, K=512, N=1536.  One 16x16 C tile per wave.
// A fragment: lane=row, e0..15 -> K 16*half + e.
// B fragment: lane=col, e0..15 -> K 16*half + e.
// ---------------------------------------------------------------------------
__global__ __launch_bounds__(256) void qkv_gemm_wmma(
    const __bf16* __restrict__ xb, const __bf16* __restrict__ wb,
    __bf16* __restrict__ qb, __bf16* __restrict__ kb_, __bf16* __restrict__ vt)
{
    const int wave = threadIdx.x >> 5;
    const int lane = threadIdx.x & 31;
    const int tile = blockIdx.x * 8 + wave;
    const int NTN  = (3 * ATT_DIM) / 16;         // 96 N-tiles
    const int mt   = tile / NTN;
    const int nt   = tile % NTN;
    if (mt >= L_SEQ / 16) return;                // uniform per wave
    const int m0   = mt * 16, n0 = nt * 16;
    const int hlf  = lane >> 4;
    const int lidx = lane & 15;
    const int kbse = hlf * 8;

    const __bf16* arow = xb + (size_t)(m0 + lidx) * IN_DIMC;
    const __bf16* brow = wb + (size_t)(n0 + lidx) * IN_DIMC;

    v8f acc = {};
    for (int k0 = 0; k0 < IN_DIMC; k0 += 32) {
        v8bf  alo = *(const v8bf*)(arow + k0 + 2 * kbse);
        v8bf  ahi = *(const v8bf*)(arow + k0 + 2 * kbse + 8);
        v16bf Af  = fuse(alo, ahi);
        v16bf Bf  = *(const v16bf*)(brow + k0 + hlf * 16);
        acc = __builtin_amdgcn_wmma_f32_16x16x32_bf16(
            false, Af, false, Bf, (short)0, acc, false, false);
    }

    // C layout: lane holds col n0+lidx, rows r + 8*hlf in acc[r]
    const int cn  = n0 + lidx;
    const int seg = cn / ATT_DIM;                 // 0=q 1=k 2=v
    const int hh  = (cn % ATT_DIM) / HD;
    const int dd  = cn % HD;
    if (seg == 2) {
        // V transposed: vt[h][d][m]; rows r are contiguous -> one 16B store
        v8bf pv;
#pragma unroll
        for (int r = 0; r < 8; ++r) pv[r] = (__bf16)acc[r];
        __bf16* dst = vt + (size_t)hh * HD * L_SEQ + (size_t)dd * L_SEQ
                         + (m0 + 8 * hlf);
        *(v8bf*)dst = pv;
    } else {
        __bf16* dst = (seg == 0 ? qb : kb_) + (size_t)hh * L_SEQ * HD + dd;
        const float sc = (seg == 0) ? 0.125f : 1.0f;   // q / sqrt(64)
#pragma unroll
        for (int r = 0; r < 8; ++r) {
            int m = m0 + r + 8 * hlf;
            dst[(size_t)m * HD] = (__bf16)(acc[r] * sc);
        }
    }
}

// ---------------------------------------------------------------------------
// K2: fused flash attention with iRPE-k contextual bias.
// 128 threads = 4 waves; wave handles one 16-row Q tile of head h.
// ---------------------------------------------------------------------------
__global__ __launch_bounds__(128) void flash_rpe_wmma(
    const __bf16* __restrict__ qb, const __bf16* __restrict__ kb_,
    const __bf16* __restrict__ vt, const __bf16* __restrict__ wrt,
    float* __restrict__ out)
{
    __shared__ int   s_pidx[128];                                    // 111 used
    __shared__ float s_tab[4][16 * NBP];                             // 16 KB
    __shared__ __attribute__((aligned(32))) __bf16 s_p[4][16 * 32];  // 4 KB

    const int tid  = threadIdx.x;
    const int wave = tid >> 5;
    const int lane = tid & 31;
    const int hlf  = lane >> 4;
    const int lidx = lane & 15;

    const int h  = blockIdx.x / (L_SEQ / 64);
    const int rg = blockIdx.x % (L_SEQ / 64);
    const int i0 = rg * 64 + wave * 16;

    // ---- piecewise-index LUT (111 entries) ----
    for (int t = tid; t < 2 * (SIDE - 1) + 1; t += 128) {
        int d = t - (SIDE - 1);
        float rpa = fabsf((float)d);
        int pi;
        if (rpa <= 1.9f) {
            pi = d;                                     // |d| in {0,1}
        } else {
            float lt = 1.9f + __logf(rpa * (1.0f / 1.9f)) * (1.9f / 2.0794415f);
            int iv = (int)rintf(lt);
            if (iv > 3) iv = 3;                         // min(round, beta) -> int
            pi = (d > 0) ? iv : -iv;
        }
        s_pidx[t] = pi + 3;                             // + BETA_INT
    }
    __syncthreads();

    // ---- Q tile in A-fragment layout (q already scaled in K1) ----
    const __bf16* qrow = qb + (size_t)h * L_SEQ * HD + (size_t)(i0 + lidx) * HD;
    v16bf aq[2];
#pragma unroll
    for (int c = 0; c < 2; ++c) {
        v8bf lo = *(const v8bf*)(qrow + c * 32 + hlf * 16);
        v8bf hi = *(const v8bf*)(qrow + c * 32 + hlf * 16 + 8);
        aq[c] = fuse(lo, hi);
    }

    // ---- contextual bias table via WMMA: tab = q_tile @ w_rpe (padded 64) ----
#pragma unroll
    for (int ntl = 0; ntl < 4; ++ntl) {
        v8f tc = {};
#pragma unroll
        for (int c = 0; c < 2; ++c) {
            v16bf Bw = *(const v16bf*)(wrt + (size_t)(ntl * 16 + lidx) * HD
                                           + c * 32 + hlf * 16);
            tc = __builtin_amdgcn_wmma_f32_16x16x32_bf16(
                false, aq[c], false, Bw, (short)0, tc, false, false);
        }
#pragma unroll
        for (int r = 0; r < 8; ++r)
            s_tab[wave][(r + 8 * hlf) * NBP + ntl * 16 + lidx] = tc[r];
    }

    // per-lane row coordinates (rows r + 8*hlf of this tile)
    int yi8[8], xi8[8];
#pragma unroll
    for (int r = 0; r < 8; ++r) {
        int m = i0 + r + 8 * hlf;
        yi8[r] = m / SIDE; xi8[r] = m % SIDE;
    }

    float m_i[8], l_i[8];
#pragma unroll
    for (int r = 0; r < 8; ++r) { m_i[r] = -1e30f; l_i[r] = 0.f; }
    v8f o0 = {}, o1 = {}, o2 = {}, o3 = {};

    const __bf16* Kh  = kb_ + (size_t)h * L_SEQ * HD;
    const __bf16* Vth = vt  + (size_t)h * HD * L_SEQ;   // [d][key]

    for (int j0 = 0; j0 < L_SEQ; j0 += 32) {
        // prefetch next key tile (global_prefetch_b8)
        if (j0 + 32 < L_SEQ) {
            __builtin_prefetch(Kh  + (size_t)(j0 + 32 + lane) * HD, 0, 1);
            __builtin_prefetch(Vth + (size_t)(2 * lane) * L_SEQ + j0 + 32, 0, 1);
            __builtin_prefetch(Vth + (size_t)(2 * lane + 1) * L_SEQ + j0 + 32, 0, 1);
        }

        // ---- S = Q.K^T for two 16-key subtiles (d=64 -> 2 chained wmma) ----
        v8f s0 = {}, s1 = {};
#pragma unroll
        for (int c = 0; c < 2; ++c) {
            v16bf B0 = *(const v16bf*)(Kh + (size_t)(j0 + lidx) * HD
                                          + c * 32 + hlf * 16);
            v16bf B1 = *(const v16bf*)(Kh + (size_t)(j0 + 16 + lidx) * HD
                                          + c * 32 + hlf * 16);
            s0 = __builtin_amdgcn_wmma_f32_16x16x32_bf16(
                false, aq[c], false, B0, (short)0, s0, false, false);
            s1 = __builtin_amdgcn_wmma_f32_16x16x32_bf16(
                false, aq[c], false, B1, (short)0, s1, false, false);
        }

        // ---- add rpe bias via bucket gather ----
        int ja = j0 + lidx, jb = ja + 16;
        int yja = ja / SIDE, xja = ja % SIDE;
        int yjb = jb / SIDE, xjb = jb % SIDE;
#pragma unroll
        for (int r = 0; r < 8; ++r) {
            int rl = r + 8 * hlf;
            int ba = s_pidx[yi8[r] - yja + 55] * 7 + s_pidx[xi8[r] - xja + 55];
            int bb = s_pidx[yi8[r] - yjb + 55] * 7 + s_pidx[xi8[r] - xjb + 55];
            s0[r] += s_tab[wave][rl * NBP + ba];
            s1[r] += s_tab[wave][rl * NBP + bb];
        }

        // ---- online softmax (row reductions across 16-lane halves) ----
        float al[8];
#pragma unroll
        for (int r = 0; r < 8; ++r) {
            float tm = fmaxf(s0[r], s1[r]);
#pragma unroll
            for (int off = 1; off < 16; off <<= 1)
                tm = fmaxf(tm, __shfl_xor(tm, off, 32));
            float mn = fmaxf(m_i[r], tm);
            al[r] = __expf(m_i[r] - mn);
            float p0 = __expf(s0[r] - mn);
            float p1 = __expf(s1[r] - mn);
            s0[r] = p0; s1[r] = p1;
            float rs = p0 + p1;
#pragma unroll
            for (int off = 1; off < 16; off <<= 1)
                rs += __shfl_xor(rs, off, 32);
            l_i[r] = l_i[r] * al[r] + rs;
            m_i[r] = mn;
        }
#pragma unroll
        for (int r = 0; r < 8; ++r) {
            o0[r] *= al[r]; o1[r] *= al[r]; o2[r] *= al[r]; o3[r] *= al[r];
        }

        // ---- relayout P: C-fragment -> LDS bf16 -> A-fragment ----
#pragma unroll
        for (int r = 0; r < 8; ++r) {
            int rl = r + 8 * hlf;
            s_p[wave][rl * 32 + lidx]      = (__bf16)s0[r];
            s_p[wave][rl * 32 + 16 + lidx] = (__bf16)s1[r];
        }
        v8bf plo = *(const v8bf*)(&s_p[wave][lidx * 32 + hlf * 16]);
        v8bf phi = *(const v8bf*)(&s_p[wave][lidx * 32 + hlf * 16 + 8]);
        v16bf AP = fuse(plo, phi);

        // ---- O += P.V  (V transposed: contiguous 32B per d-tile) ----
#pragma unroll
        for (int dt = 0; dt < 4; ++dt) {
            v16bf BV = *(const v16bf*)(Vth + (size_t)(dt * 16 + lidx) * L_SEQ
                                           + j0 + hlf * 16);
            v8f& oo = (dt == 0 ? o0 : dt == 1 ? o1 : (dt == 2 ? o2 : o3));
            oo = __builtin_amdgcn_wmma_f32_16x16x32_bf16(
                false, AP, false, BV, (short)0, oo, false, false);
        }
    }

    // ---- normalize and write out: out[m][h*64 + d], f32 ----
#pragma unroll
    for (int r = 0; r < 8; ++r) {
        int m = i0 + r + 8 * hlf;
        float inv = 1.0f / l_i[r];
        float* orow = out + (size_t)m * ATT_DIM + h * HD;
        orow[ 0 + lidx] = o0[r] * inv;
        orow[16 + lidx] = o1[r] * inv;
        orow[32 + lidx] = o2[r] * inv;
        orow[48 + lidx] = o3[r] * inv;
    }
}

// ---------------------------------------------------------------------------
extern "C" void kernel_launch(void* const* d_in, const int* in_sizes, int n_in,
                              void* d_out, int out_size, void* d_ws, size_t ws_size,
                              hipStream_t stream) {
    const float* x     = (const float*)d_in[0];   // (1, 3136, 512) f32
    const float* w_qkv = (const float*)d_in[1];   // (1536, 512)    f32
    const float* w_rpe = (const float*)d_in[2];   // (64, 50)       f32
    float* out = (float*)d_out;                   // (1, 3136, 512) f32

    // workspace layout (bf16 elements)
    const size_t NX = (size_t)L_SEQ * IN_DIMC;            // 1,605,632
    const size_t NW = (size_t)3 * ATT_DIM * IN_DIMC;      //   786,432
    const size_t NQ = (size_t)NH * L_SEQ * HD;            // 1,605,632
    __bf16* xb  = (__bf16*)d_ws;
    __bf16* wb  = xb  + NX;
    __bf16* wrt = wb  + NW;
    __bf16* qb  = wrt + (size_t)NBP * HD;
    __bf16* kb  = qb  + NQ;
    __bf16* vt  = kb  + NQ;

    pack_bf16<<<(int)((NX / 8 + 255) / 256), 256, 0, stream>>>(x, xb, (int)(NX / 8));
    pack_bf16<<<(int)((NW / 8 + 255) / 256), 256, 0, stream>>>(w_qkv, wb, (int)(NW / 8));
    pack_wrpe<<<(NBP * HD + 255) / 256, 256, 0, stream>>>(w_rpe, wrt);

    const int tiles = (L_SEQ / 16) * ((3 * ATT_DIM) / 16);   // 18816
    qkv_gemm_wmma<<<tiles / 8, 256, 0, stream>>>(xb, wb, qb, kb, vt);
    flash_rpe_wmma<<<NH * (L_SEQ / 64), 128, 0, stream>>>(qb, kb, vt, wrt, out);
}
